// PSAttn_14242111553748
// MI455X (gfx1250) — matrix operand order repaired and checked by reading, not compiled, not run on hardware
//
#include <hip/hip_runtime.h>
#include <hip/hip_bf16.h>

// ---------------- problem constants (from reference) ----------------
#define Bn   2
#define Cc   512
#define Hh   64
#define Wd   64
#define Nn   4096      // H*W
#define HuD  32
#define WuD  32
#define NuD  1024      // Hu*Wu
#define NHD  8
#define HDd  64        // head dim
#define C2   1024      // 2*C
#define TOPKD 16

typedef _Float16 f16;
typedef __attribute__((ext_vector_type(16))) _Float16 v16h;
typedef __attribute__((ext_vector_type(8)))  _Float16 v8h;
typedef __attribute__((ext_vector_type(8)))  float    v8f;

union FragH { v16h v; v8h h[2]; };

// -------- WMMA fragment loaders (CDNA5 wave32 layouts, ISA 7.12.2) --------
// A 16(M)x32(K) f16, row-major, leading dim lda:
//   lanes 0-15: M=lane, VGPR0-3 hold K=0..7, VGPR4-7 hold K=16..23
//   lanes16-31: M=lane-16, VGPR0-3 K=8..15, VGPR4-7 K=24..31
__device__ __forceinline__ v16h load_a16x32(const f16* A, int lda, int lane) {
  int r  = lane & 15;
  int ko = (lane >> 4) * 8;
  FragH f;
  f.h[0] = *(const v8h*)(A + r * lda + ko);
  f.h[1] = *(const v8h*)(A + r * lda + 16 + ko);
  return f.v;
}
// B 32(K)x16(N) f16, K-major rows, leading dim ldb:
//   lane l holds column N=l&15; elements 0..7 K=8*(l>>4)+s, 8..15 K=16+8*(l>>4)+s-8.
__device__ __forceinline__ v16h load_b32x16(const f16* B, int ldb, int lane) {
  int n  = lane & 15;
  int ko = (lane >> 4) * 8;
  v16h r;
  for (int s = 0; s < 8; ++s) {
    r[s]     = B[(ko + s) * ldb + n];
    r[8 + s] = B[(16 + ko + s) * ldb + n];
  }
  return r;
}
__device__ __forceinline__ v8f wmma_hf(v16h a, v16h b, v8f c) {
  return __builtin_amdgcn_wmma_f32_16x16x32_f16(false, a, false, b, (short)0, c,
                                                false, false);
}
// C/D fragment element (j, lane) -> M = j + 8*(lane>>4), N = lane&15.

// row reductions across the 16-lane halves of the C fragment
__device__ __forceinline__ float rowmax16(float x) {
  for (int m = 1; m < 16; m <<= 1) x = fmaxf(x, __shfl_xor(x, m, 32));
  return x;
}
__device__ __forceinline__ float rowsum16(float x) {
  for (int m = 1; m < 16; m <<= 1) x += __shfl_xor(x, m, 32);
  return x;
}

// 16x64 strip per wave, software-pipelined one K-step ahead:
// next iteration's A + 4xB fragments are in flight while current WMMAs issue,
// so v_wmma waits at high loadcnt instead of 0 (restores round-1 overlap
// while keeping 4x A-fragment reuse).
__device__ __forceinline__ void gemm_strip4(const f16* __restrict__ A, int lda,
                                            const f16* __restrict__ B, int ldb,
                                            int K, int lane, v8f acc[4]) {
  v16h af = load_a16x32(A, lda, lane);
  v16h b0 = load_b32x16(B +  0, ldb, lane);
  v16h b1 = load_b32x16(B + 16, ldb, lane);
  v16h b2 = load_b32x16(B + 32, ldb, lane);
  v16h b3 = load_b32x16(B + 48, ldb, lane);
  for (int k = 0; k < K; k += 32) {
    v16h afn = af, n0 = b0, n1 = b1, n2 = b2, n3 = b3;
    if (k + 32 < K) {
      __builtin_prefetch(A + k + 96);          // -> global_prefetch_b8
      afn = load_a16x32(A + k + 32, lda, lane);
      const f16* bp = B + (size_t)(k + 32) * ldb;
      n0 = load_b32x16(bp +  0, ldb, lane);
      n1 = load_b32x16(bp + 16, ldb, lane);
      n2 = load_b32x16(bp + 32, ldb, lane);
      n3 = load_b32x16(bp + 48, ldb, lane);
    }
    acc[0] = wmma_hf(af, b0, acc[0]);
    acc[1] = wmma_hf(af, b1, acc[1]);
    acc[2] = wmma_hf(af, b2, acc[2]);
    acc[3] = wmma_hf(af, b3, acc[3]);
    af = afn; b0 = n0; b1 = n1; b2 = n2; b3 = n3;
  }
}

// ============================ pack kernels ============================
__global__ void ps_zero(float* p, int n) {
  int i = blockIdx.x * blockDim.x + threadIdx.x;
  if (i < n) p[i] = 0.0f;
}
// W (rows x cols) f32 row-major -> out (cols x rows) f16 (K-major for WMMA B)
__global__ void ps_transpose(const float* __restrict__ W, f16* __restrict__ out,
                             int rows, int cols) {
  int i = blockIdx.x * blockDim.x + threadIdx.x;
  if (i >= rows * cols) return;
  int o = i / cols, k = i % cols;
  out[k * rows + o] = (f16)W[i];
}
// x (B,C,N) f32 -> xT (B,N,C) f16
__global__ void ps_pack(const float* __restrict__ src, f16* __restrict__ dst,
                        int C, int N) {
  long i = (long)blockIdx.x * blockDim.x + threadIdx.x;
  if (i >= (long)Bn * C * N) return;
  int n = (int)(i % N);
  int c = (int)((i / N) % C);
  int b = (int)(i / ((long)N * C));
  dst[((long)b * N + n) * C + c] = (f16)src[i];
}
// heat[b, mu] = mean over channels of 2x2-avg-pooled x
__global__ void ps_heat(const float* __restrict__ x, float* __restrict__ heat) {
  int i = blockIdx.x * blockDim.x + threadIdx.x;
  if (i >= Bn * NuD) return;
  int b = i / NuD, mu = i % NuD;
  int hu = mu / WuD, wu = mu % WuD;
  float s = 0.f;
  for (int c = 0; c < Cc; ++c) {
    const float* p = x + ((long)(b * Cc + c) * Nn) + (2 * hu) * Wd + 2 * wu;
    s += p[0] + p[1] + p[Wd] + p[Wd + 1];
  }
  heat[i] = s * (1.0f / (Cc * 4));
}

// ============================ GEMM kernels ============================
// q = xT @ q_w^T + q_b  ->  f16 [B*N][C]
__global__ __launch_bounds__(256) void ps_gemm_q(const f16* __restrict__ xT,
                                                 const f16* __restrict__ qwT,
                                                 const float* __restrict__ qb,
                                                 f16* __restrict__ q) {
  int lane = threadIdx.x & 31;
  int tile = blockIdx.x * 8 + (threadIdx.x >> 5);
  const int NT4 = Cc / 64;                       // 8 strips of 64 cols
  if (tile >= (Bn * Nn / 16) * NT4) return;
  int nt4 = tile % NT4, mt = tile / NT4;
  v8f acc[4]; { v8f z = {}; for (int i = 0; i < 4; ++i) acc[i] = z; }
  gemm_strip4(xT + (long)mt * 16 * Cc, Cc, qwT + nt4 * 64, Cc, Cc, lane, acc);
  int hi = lane >> 4;
  for (int nt = 0; nt < 4; ++nt) {
    int nn = nt4 * 64 + nt * 16 + (lane & 15);
    float bias = qb[nn];
    for (int j = 0; j < 8; ++j)
      q[((long)mt * 16 + j + 8 * hi) * Cc + nn] = (f16)(acc[nt][j] + bias);
  }
}
// kv on upper feat -> k stored d-major kT[b][h][d][m], v f16 [b][h][m][d],
// and v f32 channel-major [b][c][m] for the depthwise PE conv.
__global__ __launch_bounds__(256) void ps_gemm_kv(const f16* __restrict__ uT,
                                                  const f16* __restrict__ kvwT,
                                                  const float* __restrict__ kvb,
                                                  f16* __restrict__ kT,
                                                  f16* __restrict__ v,
                                                  float* __restrict__ vchan) {
  int lane = threadIdx.x & 31;
  int tile = blockIdx.x * 8 + (threadIdx.x >> 5);
  const int NT4 = C2 / 64;                       // 16
  if (tile >= (Bn * NuD / 16) * NT4) return;
  int nt4 = tile % NT4, mt = tile / NT4;
  v8f acc[4]; { v8f z = {}; for (int i = 0; i < 4; ++i) acc[i] = z; }
  gemm_strip4(uT + (long)mt * 16 * Cc, Cc, kvwT + nt4 * 64, C2, Cc, lane, acc);
  int hi = lane >> 4;
  for (int nt = 0; nt < 4; ++nt) {
    int o = nt4 * 64 + nt * 16 + (lane & 15);    // output channel 0..1023
    int h = o >> 7, jj = o & 127;
    float bias = kvb[o];
    for (int j = 0; j < 8; ++j) {
      int row = mt * 16 + j + 8 * hi;            // b*Nu + m
      int b = row / NuD, m = row % NuD;
      float val = acc[nt][j] + bias;
      if (jj < HDd) {
        kT[(((long)b * NHD + h) * HDd + jj) * NuD + m] = (f16)val;
      } else {
        int d = jj - HDd;
        v[(((long)b * NHD + h) * NuD + m) * HDd + d] = (f16)val;
        vchan[((long)b * Cc + h * HDd + d) * NuD + m] = val;
      }
    }
  }
}
// f_kv on full-res x -> f_k / f_v f16 [b][h][n][d] (row-major for gathering)
__global__ __launch_bounds__(256) void ps_gemm_fkv(const f16* __restrict__ xT,
                                                   const f16* __restrict__ kvwT,
                                                   const float* __restrict__ kvb,
                                                   f16* __restrict__ fk,
                                                   f16* __restrict__ fv) {
  int lane = threadIdx.x & 31;
  int tile = blockIdx.x * 8 + (threadIdx.x >> 5);
  const int NT4 = C2 / 64;
  if (tile >= (Bn * Nn / 16) * NT4) return;
  int nt4 = tile % NT4, mt = tile / NT4;
  v8f acc[4]; { v8f z = {}; for (int i = 0; i < 4; ++i) acc[i] = z; }
  gemm_strip4(xT + (long)mt * 16 * Cc, Cc, kvwT + nt4 * 64, C2, Cc, lane, acc);
  int hi = lane >> 4;
  for (int nt = 0; nt < 4; ++nt) {
    int o = nt4 * 64 + nt * 16 + (lane & 15);
    int h = o >> 7, jj = o & 127;
    float bias = kvb[o];
    for (int j = 0; j < 8; ++j) {
      long row = (long)mt * 16 + j + 8 * hi;     // b*N + n
      long b = row / Nn, n = row % Nn;
      float val = acc[nt][j] + bias;
      if (jj < HDd) fk[((b * NHD + h) * Nn + n) * HDd + jj] = (f16)val;
      else          fv[((b * NHD + h) * Nn + n) * HDd + (jj - HDd)] = (f16)val;
    }
  }
}

// ===================== coarse attention (flash-style) =====================
// one wave per (b,h, 16-row block of n); two passes over m (recompute sim).
__global__ __launch_bounds__(32) void ps_attn(const f16* __restrict__ q,
                                              const f16* __restrict__ kT,
                                              const f16* __restrict__ v,
                                              float* __restrict__ colsum,
                                              float* __restrict__ coarse) {
  __shared__ __align__(32) f16 sP[16 * 32];
  int lane = threadIdx.x & 31;
  int t = blockIdx.x;
  int mt16 = t & 255;                 // Nn/16 = 256
  int h = (t >> 8) & 7;
  int b = t >> 11;
  int bh = b * NHD + h;
  const f16* Aq = q + ((long)b * Nn + mt16 * 16) * Cc + h * HDd;  // lda=Cc
  const f16* Bk = kT + (long)bh * HDd * NuD;                      // [64][1024]
  const f16* Bv = v + (long)bh * NuD * HDd;                       // [1024][64]
  const float scale = 0.125f;         // 64^-0.5
  int hi = lane >> 4;

  // q fragments are loop-invariant: load once
  v16h aq0 = load_a16x32(Aq, Cc, lane);
  v16h aq1 = load_a16x32(Aq + 32, Cc, lane);

  float rm[8], rs[8];
  for (int j = 0; j < 8; ++j) { rm[j] = -1e30f; rs[j] = 0.f; }

  // pass 1: online row max / sum-exp + column sums of sim for global_sim,
  // with B fragments pipelined one m-tile ahead (softmax VALU covers latency)
  v16h bk0 = load_b32x16(Bk, NuD, lane);
  v16h bk1 = load_b32x16(Bk + 32 * NuD, NuD, lane);
  for (int mt = 0; mt < NuD / 16; ++mt) {
    v16h nb0 = bk0, nb1 = bk1;
    if (mt + 1 < NuD / 16) {
      nb0 = load_b32x16(Bk + (mt + 1) * 16, NuD, lane);
      nb1 = load_b32x16(Bk + 32 * NuD + (mt + 1) * 16, NuD, lane);
    }
    v8f s = {};
    s = wmma_hf(aq0, bk0, s);
    s = wmma_hf(aq1, bk1, s);
    float col = 0.f;
    for (int j = 0; j < 8; ++j) {
      float val = s[j] * scale;
      col += val;
      float tm = rowmax16(val);
      float nm = fmaxf(rm[j], tm);
      float e = __expf(val - nm);
      rs[j] = rs[j] * __expf(rm[j] - nm) + rowsum16(e);
      rm[j] = nm;
    }
    col += __shfl_xor(col, 16, 32);   // add the other half's 8 rows
    if (lane < 16) atomicAdd(&colsum[bh * NuD + mt * 16 + lane], col);
    bk0 = nb0; bk1 = nb1;
  }
  float rinv[8];
  for (int j = 0; j < 8; ++j) rinv[j] = 1.0f / rs[j];

  // pass 2: recompute sim, normalize to f16 P via LDS, accumulate P @ V
  v8f acc[4]; { v8f z = {}; for (int i = 0; i < 4; ++i) acc[i] = z; }
  for (int mc = 0; mc < NuD / 32; ++mc) {
    for (int half = 0; half < 2; ++half) {
      int mt = mc * 2 + half;
      v8f s = {};
      s = wmma_hf(aq0, load_b32x16(Bk + mt * 16, NuD, lane), s);
      s = wmma_hf(aq1, load_b32x16(Bk + 32 * NuD + mt * 16, NuD, lane), s);
      for (int j = 0; j < 8; ++j) {
        float p = __expf(s[j] * scale - rm[j]) * rinv[j];
        sP[(j + 8 * hi) * 32 + half * 16 + (lane & 15)] = (f16)p;
      }
    }
    __syncthreads();                  // single wave: LDS is in-order anyway
    v16h ap = load_a16x32(sP, 32, lane);
    for (int nt = 0; nt < 4; ++nt)
      acc[nt] = wmma_hf(ap,
                        load_b32x16(Bv + (long)mc * 32 * HDd + nt * 16, HDd, lane),
                        acc[nt]);
    __syncthreads();
  }
  for (int nt = 0; nt < 4; ++nt)
    for (int j = 0; j < 8; ++j)
      coarse[((long)bh * Nn + mt16 * 16 + j + 8 * hi) * HDd + nt * 16 + (lane & 15)]
          = acc[nt][j];
}

// ===================== top-k indices (softmax is monotonic) =====================
__global__ __launch_bounds__(32) void ps_topk(const float* __restrict__ colsum,
                                              const float* __restrict__ heat,
                                              const float* __restrict__ gumbel,
                                              int* __restrict__ idxfull) {
  int lane = threadIdx.x;
  int bh = blockIdx.x;
  int b = bh / NHD;
  float vals[32];
  for (int i = 0; i < 32; ++i) {
    int m = i * 32 + lane;
    vals[i] = colsum[bh * NuD + m] * (1.0f / Nn) * heat[b * NuD + m]
            + gumbel[(long)bh * NuD + m];
  }
  unsigned taken = 0u;
  for (int t = 0; t < TOPKD; ++t) {
    float bv = -1e30f; int bi = 0;
    for (int i = 0; i < 32; ++i)
      if (!(taken & (1u << i)) && vals[i] > bv) { bv = vals[i]; bi = i; }
    int bm = bi * 32 + lane;
    for (int m = 16; m >= 1; m >>= 1) {
      float ov = __shfl_xor(bv, m, 32);
      int   om = __shfl_xor(bm, m, 32);
      if (ov > bv || (ov == bv && om < bm)) { bv = ov; bm = om; }
    }
    if ((bm & 31) == lane) taken |= 1u << (bm >> 5);
    if (lane == 0) {
      int hu = bm / WuD, wu = bm % WuD;
      int h2 = hu * 2, w2 = wu * 2;
      idxfull[bh * 64 + 0 * 16 + t] = (h2 + 0) * Wd + (w2 + 0);
      idxfull[bh * 64 + 1 * 16 + t] = (h2 + 0) * Wd + (w2 + 1);
      idxfull[bh * 64 + 2 * 16 + t] = (h2 + 1) * Wd + (w2 + 0);
      idxfull[bh * 64 + 3 * 16 + t] = (h2 + 1) * Wd + (w2 + 1);
    }
  }
}

// ===== fine attention + gated fusion; writes x_out n-major f32 =====
__global__ __launch_bounds__(32) void ps_fine(const f16* __restrict__ q,
                                              const f16* __restrict__ fk,
                                              const f16* __restrict__ fv,
                                              const int* __restrict__ idxfull,
                                              const float* __restrict__ coarse,
                                              const f16* __restrict__ gwT,
                                              const float* __restrict__ gb,
                                              float* __restrict__ xout) {
  __shared__ __align__(32) f16 sK[64 * 64];   // gathered k, d-major
  __shared__ __align__(32) f16 sV[64 * 64];   // gathered v, m-major
  __shared__ __align__(32) f16 sP[16 * 64];
  __shared__ __align__(32) f16 sC[16 * 64];   // coarse tile f16 (gate A-frag)
  __shared__ __align__(32) f16 sR[16 * 64];   // refined tile f16
  int lane = threadIdx.x & 31;
  int t = blockIdx.x;
  int mt16 = t & 255;
  int h = (t >> 8) & 7;
  int b = t >> 11;
  int bh = b * NHD + h;
  int row0 = mt16 * 16;
  int hi = lane >> 4;
  const float scale = 0.125f;

  // gather 64 rows of f_k / f_v (one wave, 2 rows per lane)
  for (int r = lane; r < 64; r += 32) {
    int idx = idxfull[bh * 64 + r];
    const f16* sk = fk + ((long)bh * Nn + idx) * HDd;
    const f16* sv = fv + ((long)bh * Nn + idx) * HDd;
    for (int d = 0; d < HDd; ++d) sK[d * 64 + r] = sk[d];   // transpose store
    for (int c8 = 0; c8 < 8; ++c8)
      *(v8h*)(sV + r * 64 + c8 * 8) = *(const v8h*)(sv + c8 * 8);
  }
  for (int i = lane; i < 16 * 64; i += 32)
    sC[i] = (f16)coarse[((long)bh * Nn + row0 + (i >> 6)) * HDd + (i & 63)];
  __syncthreads();

  // fine sim (16 x 64), softmax over the 64 gathered keys
  const f16* Aq = q + ((long)b * Nn + row0) * Cc + h * HDd;
  v16h aq0 = load_a16x32(Aq, Cc, lane);
  v16h aq1 = load_a16x32(Aq + 32, Cc, lane);
  v8f s[4]; { v8f z = {}; for (int i = 0; i < 4; ++i) s[i] = z; }
  for (int mt = 0; mt < 4; ++mt) {
    s[mt] = wmma_hf(aq0, load_b32x16(sK + mt * 16, 64, lane), s[mt]);
    s[mt] = wmma_hf(aq1, load_b32x16(sK + 32 * 64 + mt * 16, 64, lane), s[mt]);
  }
  float rm[8], rs[8];
  for (int j = 0; j < 8; ++j) {
    float m = -1e30f;
    for (int mt = 0; mt < 4; ++mt) m = fmaxf(m, rowmax16(s[mt][j] * scale));
    rm[j] = m;
    float su = 0.f;
    for (int mt = 0; mt < 4; ++mt) su += rowsum16(__expf(s[mt][j] * scale - m));
    rs[j] = 1.0f / su;
  }
  for (int mt = 0; mt < 4; ++mt)
    for (int j = 0; j < 8; ++j)
      sP[(j + 8 * hi) * 64 + mt * 16 + (lane & 15)]
          = (f16)(__expf(s[mt][j] * scale - rm[j]) * rs[j]);
  __syncthreads();

  // refined = P @ V_topk
  v8f ref[4]; { v8f z = {}; for (int i = 0; i < 4; ++i) ref[i] = z; }
  {
    v16h ap0 = load_a16x32(sP, 64, lane);
    v16h ap1 = load_a16x32(sP + 32, 64, lane);
    for (int nt = 0; nt < 4; ++nt) {
      ref[nt] = wmma_hf(ap0, load_b32x16(sV + nt * 16, 64, lane), ref[nt]);
      ref[nt] = wmma_hf(ap1, load_b32x16(sV + 32 * 64 + nt * 16, 64, lane), ref[nt]);
    }
  }
  for (int nt = 0; nt < 4; ++nt)
    for (int j = 0; j < 8; ++j)
      sR[(j + 8 * hi) * 64 + nt * 16 + (lane & 15)] = (f16)ref[nt][j];
  __syncthreads();

  // gate = sigmoid([coarse|refined] @ gate_w^T + b); combine; write n-major
  v16h ac0 = load_a16x32(sC, 64, lane);
  v16h ac1 = load_a16x32(sC + 32, 64, lane);
  v16h ar0 = load_a16x32(sR, 64, lane);
  v16h ar1 = load_a16x32(sR + 32, 64, lane);
  for (int ot = 0; ot < 4; ++ot) {
    v8f g = {};
    g = wmma_hf(ac0, load_b32x16(gwT + 0 * 32 * 64 + ot * 16, 64, lane), g);
    g = wmma_hf(ac1, load_b32x16(gwT + 1 * 32 * 64 + ot * 16, 64, lane), g);
    g = wmma_hf(ar0, load_b32x16(gwT + 2 * 32 * 64 + ot * 16, 64, lane), g);
    g = wmma_hf(ar1, load_b32x16(gwT + 3 * 32 * 64 + ot * 16, 64, lane), g);
    int o = ot * 16 + (lane & 15);
    float bias = gb[o];
    for (int j = 0; j < 8; ++j) {
      int M = j + 8 * hi;
      float co = coarse[((long)bh * Nn + row0 + M) * HDd + o];
      float ga = 1.0f / (1.0f + __expf(-(g[j] + bias)));
      float ov = ga * ref[ot][j] + (1.0f - ga) * co;
      xout[((long)b * Nn + row0 + M) * Cc + h * HDd + o] = ov;
    }
  }
}

// ============ depthwise 7x7 conv (pad 3) on v, channel-major ============
__global__ void ps_dwconv(const float* __restrict__ vchan,
                          const float* __restrict__ pw,
                          const float* __restrict__ pb,
                          float* __restrict__ vpe) {
  int i = blockIdx.x * blockDim.x + threadIdx.x;
  if (i >= Bn * Cc * NuD) return;
  int mu = i % NuD;
  int c = (i / NuD) % Cc;
  int hu = mu / WuD, wu = mu % WuD;
  const float* src = vchan + (long)(i / NuD) * NuD;
  const float* w = pw + c * 49;
  float s = pb[c];
  for (int kh = 0; kh < 7; ++kh) {
    int y = hu + kh - 3;
    if (y < 0 || y >= HuD) continue;
    for (int kw = 0; kw < 7; ++kw) {
      int xx = wu + kw - 3;
      if (xx < 0 || xx >= WuD) continue;
      s += src[y * WuD + xx] * w[kh * 7 + kw];
    }
  }
  vpe[i] = s;
}

// ==== bilinear x2 upsample (half-pixel) + add x_out -> f16 proj input ====
__global__ void ps_resize_add(const float* __restrict__ vpe,
                              const float* __restrict__ xout,
                              f16* __restrict__ aT) {
  long i = (long)blockIdx.x * blockDim.x + threadIdx.x;
  if (i >= (long)Bn * Nn * Cc) return;
  int c = (int)(i % Cc);
  long bn = i / Cc;
  int n = (int)(bn % Nn);
  int b = (int)(bn / Nn);
  int hh = n / Wd, ww = n % Wd;
  float fy = hh * 0.5f - 0.25f, fx = ww * 0.5f - 0.25f;
  int y0 = (int)floorf(fy), x0 = (int)floorf(fx);
  float wy = fy - (float)y0, wx = fx - (float)x0;
  int y0c = min(max(y0, 0), HuD - 1), y1c = min(max(y0 + 1, 0), HuD - 1);
  int x0c = min(max(x0, 0), WuD - 1), x1c = min(max(x0 + 1, 0), WuD - 1);
  const float* src = vpe + ((long)b * Cc + c) * NuD;
  float s = (1.f - wy) * ((1.f - wx) * src[y0c * WuD + x0c] + wx * src[y0c * WuD + x1c])
          +        wy  * ((1.f - wx) * src[y1c * WuD + x0c] + wx * src[y1c * WuD + x1c]);
  aT[i] = (f16)(xout[i] + s);
}

// ================= final projection GEMM -> fp32 NCHW output =================
__global__ __launch_bounds__(256) void ps_gemm_proj(const f16* __restrict__ aT,
                                                    const f16* __restrict__ pwT,
                                                    const float* __restrict__ pb,
                                                    float* __restrict__ out) {
  int lane = threadIdx.x & 31;
  int tile = blockIdx.x * 8 + (threadIdx.x >> 5);
  const int NT4 = Cc / 64;
  if (tile >= (Bn * Nn / 16) * NT4) return;
  int nt4 = tile % NT4, mt = tile / NT4;
  v8f acc[4]; { v8f z = {}; for (int i = 0; i < 4; ++i) acc[i] = z; }
  gemm_strip4(aT + (long)mt * 16 * Cc, Cc, pwT + nt4 * 64, Cc, Cc, lane, acc);
  int hi = lane >> 4;
  for (int nt = 0; nt < 4; ++nt) {
    int o = nt4 * 64 + nt * 16 + (lane & 15);
    float bias = pb[o];
    for (int j = 0; j < 8; ++j) {
      long row = (long)mt * 16 + j + 8 * hi;
      long b = row / Nn, n = row % Nn;
      out[(b * Cc + o) * Nn + n] = acc[nt][j] + bias;
    }
  }
}

// ================================ host ================================
extern "C" void kernel_launch(void* const* d_in, const int* in_sizes, int n_in,
                              void* d_out, int out_size, void* d_ws, size_t ws_size,
                              hipStream_t stream) {
  (void)in_sizes; (void)n_in; (void)out_size; (void)ws_size;
  const float* x      = (const float*)d_in[0];
  const float* upper  = (const float*)d_in[1];
  const float* gumbel = (const float*)d_in[2];
  const float* q_w    = (const float*)d_in[3];
  const float* q_b    = (const float*)d_in[4];
  const float* kv_w   = (const float*)d_in[5];
  const float* kv_b   = (const float*)d_in[6];
  const float* proj_w = (const float*)d_in[7];
  const float* proj_b = (const float*)d_in[8];
  const float* pe_w   = (const float*)d_in[9];
  const float* pe_b   = (const float*)d_in[10];
  const float* gate_w = (const float*)d_in[11];
  const float* gate_b = (const float*)d_in[12];

  char* ws = (char*)d_ws;
  size_t off = 0;
  auto alloc = [&](size_t bytes) -> char* {
    char* p = ws + off;
    off = (off + bytes + 255) & ~(size_t)255;
    return p;
  };
  f16*   qwT    = (f16*)  alloc((size_t)Cc * Cc * 2);
  f16*   kvwT   = (f16*)  alloc((size_t)Cc * C2 * 2);
  f16*   pwT    = (f16*)  alloc((size_t)Cc * Cc * 2);
  f16*   gwT    = (f16*)  alloc((size_t)128 * 64 * 2);
  f16*   xT     = (f16*)  alloc((size_t)Bn * Nn * Cc * 2);
  f16*   uT     = (f16*)  alloc((size_t)Bn * NuD * Cc * 2);
  float* heat   = (float*)alloc((size_t)Bn * NuD * 4);
  f16*   qb_    = (f16*)  alloc((size_t)Bn * Nn * Cc * 2);
  f16*   kTb    = (f16*)  alloc((size_t)Bn * NHD * HDd * NuD * 2);
  f16*   vb     = (f16*)  alloc((size_t)Bn * NHD * NuD * HDd * 2);
  float* vchan  = (float*)alloc((size_t)Bn * Cc * NuD * 4);
  f16*   fk     = (f16*)  alloc((size_t)Bn * NHD * Nn * HDd * 2);
  f16*   fv     = (f16*)  alloc((size_t)Bn * NHD * Nn * HDd * 2);
  float* colsum = (float*)alloc((size_t)Bn * NHD * NuD * 4);
  float* coarse = (float*)alloc((size_t)Bn * NHD * Nn * HDd * 4);
  int*   idxf   = (int*)  alloc((size_t)Bn * NHD * 64 * 4);
  float* xout   = (float*)alloc((size_t)Bn * Nn * Cc * 4);
  float* vpe    = (float*)alloc((size_t)Bn * Cc * NuD * 4);
  f16*   aT     = (f16*)  alloc((size_t)Bn * Nn * Cc * 2);

  const int TPB = 256;
  auto cdiv = [](long a, long b) { return (int)((a + b - 1) / b); };

  ps_transpose<<<cdiv((long)Cc * Cc, TPB), TPB, 0, stream>>>(q_w, qwT, Cc, Cc);
  ps_transpose<<<cdiv((long)C2 * Cc, TPB), TPB, 0, stream>>>(kv_w, kvwT, C2, Cc);
  ps_transpose<<<cdiv((long)Cc * Cc, TPB), TPB, 0, stream>>>(proj_w, pwT, Cc, Cc);
  ps_transpose<<<cdiv(64L * 128, TPB), TPB, 0, stream>>>(gate_w, gwT, 64, 128);
  ps_pack<<<cdiv((long)Bn * Cc * Nn, TPB), TPB, 0, stream>>>(x, xT, Cc, Nn);
  ps_pack<<<cdiv((long)Bn * Cc * NuD, TPB), TPB, 0, stream>>>(upper, uT, Cc, NuD);
  ps_heat<<<cdiv((long)Bn * NuD, TPB), TPB, 0, stream>>>(x, heat);
  ps_zero<<<cdiv((long)Bn * NHD * NuD, TPB), TPB, 0, stream>>>(colsum, Bn * NHD * NuD);

  ps_gemm_q  <<<(Bn * Nn / 16) * (Cc / 64) / 8, 256, 0, stream>>>(xT, qwT, q_b, qb_);
  ps_gemm_kv <<<(Bn * NuD / 16) * (C2 / 64) / 8, 256, 0, stream>>>(uT, kvwT, kv_b, kTb, vb, vchan);
  ps_gemm_fkv<<<(Bn * Nn / 16) * (C2 / 64) / 8, 256, 0, stream>>>(xT, kvwT, kv_b, fk, fv);

  ps_attn<<<Bn * NHD * (Nn / 16), 32, 0, stream>>>(qb_, kTb, vb, colsum, coarse);
  ps_topk<<<Bn * NHD, 32, 0, stream>>>(colsum, heat, gumbel, idxf);
  ps_fine<<<Bn * NHD * (Nn / 16), 32, 0, stream>>>(qb_, fk, fv, idxf, coarse, gwT, gate_b, xout);

  ps_dwconv<<<cdiv((long)Bn * Cc * NuD, TPB), TPB, 0, stream>>>(vchan, pe_w, pe_b, vpe);
  ps_resize_add<<<cdiv((long)Bn * Nn * Cc, TPB), TPB, 0, stream>>>(vpe, xout, aT);
  ps_gemm_proj<<<(Bn * Nn / 16) * (Cc / 64) / 8, 256, 0, stream>>>(aT, pwT, proj_b, (float*)d_out);
}
